// EfficientCausalAttention_45853070852386
// MI455X (gfx1250) — compile-verified, matrix operand order checked by reading, not run on hardware
//
#include <hip/hip_runtime.h>

#define DMODEL 1024
#define NHEAD  16
#define DK     64
#define WIN    512
#define TSEQ   2048
#define BATCH  2
#define NTOK   (BATCH * TSEQ)
#define D3     (3 * DMODEL)
#define BPAD   40   // LDS row stride (elements) for a 32-wide k-slice; kills bank conflicts

typedef __attribute__((ext_vector_type(16))) __bf16 v16bf;
typedef __attribute__((ext_vector_type(8)))  __bf16 v8bf;
typedef __attribute__((ext_vector_type(8)))  float  v8f;

#define WMMA_BF16(A, B, C) \
  __builtin_amdgcn_wmma_f32_16x16x32_bf16(false, (A), false, (B), (short)0, (C), false, false)

static __device__ __forceinline__ v16bf frag_cat(v8bf a, v8bf b) {
  v16bf r;
#pragma unroll
  for (int i = 0; i < 8; ++i) { r[i] = a[i]; r[i + 8] = b[i]; }
  return r;
}

// 16x32 bf16 WMMA fragment: per ISA layout, lane holds its row/column with
// k-chunks at k0 = kstep + (lane>>4)*8 and k0+16.
static __device__ __forceinline__ v16bf load_frag(const __bf16* rowptr, int k0) {
  v8bf lo = *(const v8bf*)(rowptr + k0);
  v8bf hi = *(const v8bf*)(rowptr + k0 + 16);
  return frag_cat(lo, hi);
}

// Async global->LDS copy, 16B per lane, tracked by ASYNCcnt.
// ldsoff = low 32 bits of the generic pointer to the shared destination
// (flat LDS addresses carry the wave-relative LDS byte address in [31:0]).
static __device__ __forceinline__ void async_ld_b128(unsigned ldsoff, const void* g) {
  asm volatile("global_load_async_to_lds_b128 %0, %1, off"
               :: "v"(ldsoff), "v"(g) : "memory");
}
static __device__ __forceinline__ void async_wait0() {
  asm volatile("s_wait_asynccnt 0x0" ::: "memory");
}
static __device__ __forceinline__ void async_wait1() {   // oldest of <=2 in flight done
  asm volatile("s_wait_asynccnt 0x1" ::: "memory");
}
static __device__ __forceinline__ unsigned lds_off(const void* p) {
  return (unsigned)(unsigned long long)p;
}

// Double-buffered, async-staged GEMM mainloop: A rows direct from global,
// B k-slices (64 cols x 32 k) streamed into LDS overlapping compute.
static __device__ __forceinline__ void gemm_mainloop(
    const __bf16* __restrict__ arow, const __bf16* __restrict__ gsrc,
    __bf16* bstage0, __bf16* bstage1, unsigned soff0, unsigned soff1,
    int half, int l16, v8f acc[4]) {
  async_ld_b128(soff0, gsrc);                       // fill buffer 0 with k=0
  int cur = 0;
  for (int k = 0; k < DMODEL; k += 32) {
    if (k + 32 < DMODEL) {
      async_ld_b128(cur ? soff0 : soff1, gsrc + (k + 32));   // prefetch next tile
      async_wait1();                                         // current tile done
    } else {
      async_wait0();
    }
    __syncthreads();                                // current tile visible to all waves
    const __bf16* buf = cur ? bstage1 : bstage0;
    v16bf Af  = load_frag(arow, k + half * 8);
    v16bf Bf0 = load_frag(buf + (0 * 16 + l16) * BPAD, half * 8);
    v16bf Bf1 = load_frag(buf + (1 * 16 + l16) * BPAD, half * 8);
    v16bf Bf2 = load_frag(buf + (2 * 16 + l16) * BPAD, half * 8);
    v16bf Bf3 = load_frag(buf + (3 * 16 + l16) * BPAD, half * 8);
    acc[0] = WMMA_BF16(Af, Bf0, acc[0]);
    acc[1] = WMMA_BF16(Af, Bf1, acc[1]);
    acc[2] = WMMA_BF16(Af, Bf2, acc[2]);
    acc[3] = WMMA_BF16(Af, Bf3, acc[3]);
    __syncthreads();                                // all reads done before re-stage
    cur ^= 1;
  }
}

// ---------------- stage 0: conversions ----------------
__global__ void k_cvt_x(const float* __restrict__ x, __bf16* __restrict__ xb, int n) {
  int i = blockIdx.x * blockDim.x + threadIdx.x;
  if (i < n) xb[i] = (__bf16)x[i];
}

// W [rows, cols] f32  ->  WT [cols, rows] bf16
__global__ void k_transpose(const float* __restrict__ W, __bf16* __restrict__ WT,
                            int rows, int cols) {
  int i = blockIdx.x * blockDim.x + threadIdx.x;
  if (i < rows * cols) {
    int r = i / cols, c = i % cols;
    WT[(long)c * rows + r] = (__bf16)W[i];
  }
}

// ---------------- stage 1: QKV projection ----------------
// Block = 128 rows x 64 cols. 8 waves, each wave: one 16-row M-tile x 4 N-tiles.
__global__ void __launch_bounds__(256) k_qkv_gemm(
    const __bf16* __restrict__ xb, const __bf16* __restrict__ wT,
    __bf16* __restrict__ Qs, __bf16* __restrict__ Ks, __bf16* __restrict__ Vt) {
  __shared__ __align__(16) __bf16 bstage[2][64 * BPAD];

  const int NB = D3 / 64;  // 48 column-blocks
  int mb = blockIdx.x / NB, nb = blockIdx.x % NB;
  int m0 = mb * 128, n0 = nb * 64;
  int wave = threadIdx.x >> 5, lane = threadIdx.x & 31;
  int half = lane >> 4, l16 = lane & 15;

  // staging: thread t -> row r = t/4 of 64, chunk c = t%4 (8 bf16 = 16B)
  int r = threadIdx.x >> 2, c = threadIdx.x & 3;
  const __bf16* gsrc = wT + (long)(n0 + r) * DMODEL + c * 8;
  unsigned soff0 = lds_off(&bstage[0][r * BPAD + c * 8]);
  unsigned soff1 = lds_off(&bstage[1][r * BPAD + c * 8]);

  const __bf16* arow = xb + (long)(m0 + wave * 16 + l16) * DMODEL;

  v8f acc[4] = {};
  gemm_mainloop(arow, gsrc, &bstage[0][0], &bstage[1][0], soff0, soff1, half, l16, acc);

  // C layout: lane -> column l16 (within tile j), vgpr v -> row half*8+v
  int which = n0 / DMODEL;                 // uniform across block (n0 64-aligned)
  int h = (n0 % DMODEL) / DK;              // uniform across block
  int tbase = m0 + wave * 16 + half * 8;
  int bb = tbase / TSEQ, tl = tbase % TSEQ;
  long qkbase = ((long)(bb * NHEAD + h) * TSEQ + tl) * DK;

  if (which == 0) {
#pragma unroll
    for (int j = 0; j < 4; ++j)
#pragma unroll
      for (int v = 0; v < 8; ++v)
        Qs[qkbase + (long)v * DK + j * 16 + l16] = (__bf16)(acc[j][v] * 0.125f);
  } else if (which == 1) {
#pragma unroll
    for (int j = 0; j < 4; ++j)
#pragma unroll
      for (int v = 0; v < 8; ++v)
        Ks[qkbase + (long)v * DK + j * 16 + l16] = (__bf16)acc[j][v];
  } else {
#pragma unroll
    for (int j = 0; j < 4; ++j) {
      int d = j * 16 + l16;
      v8bf pk;
#pragma unroll
      for (int v = 0; v < 8; ++v) pk[v] = (__bf16)acc[j][v];
      *(v8bf*)(Vt + ((long)(bb * NHEAD + h) * DK + d) * TSEQ + tl) = pk;
    }
  }
}

// ---------------- stage 2: windowed flash attention ----------------
__global__ void __launch_bounds__(256) k_attn(
    const __bf16* __restrict__ Qs, const __bf16* __restrict__ Ks,
    const __bf16* __restrict__ Vt, __bf16* __restrict__ AO) {
  __shared__ __align__(16) __bf16 lds[8][16 * BPAD];

  int wave = threadIdx.x >> 5, lane = threadIdx.x & 31;
  int half = lane >> 4, l16 = lane & 15;
  int g = blockIdx.x * 8 + wave;            // q-tile id, 0..4095
  int qt = g % (TSEQ / 16);
  int bh = g / (TSEQ / 16);
  int qbase = qt * 16;

  const __bf16* Qp = Qs + (long)bh * TSEQ * DK;
  const __bf16* Kp = Ks + (long)bh * TSEQ * DK;
  const __bf16* Vp = Vt + (long)bh * DK * TSEQ;

  v16bf QA0 = load_frag(Qp + (long)(qbase + l16) * DK, 0 + half * 8);
  v16bf QA1 = load_frag(Qp + (long)(qbase + l16) * DK, 32 + half * 8);

  v8f O0 = {}, O1 = {}, O2 = {}, O3 = {};
  float rm[8], rl[8];
#pragma unroll
  for (int v = 0; v < 8; ++v) { rm[v] = -3.0e38f; rl[v] = 0.f; }

  int kmin = qbase - (WIN - 1);
  if (kmin < 0) kmin = 0;
  kmin &= ~31;
  __bf16* P = &lds[wave][0];

  for (int kt = kmin; kt < qbase + 16; kt += 32) {
    __builtin_prefetch(Kp + (long)(kt + 32 + l16) * DK, 0, 1);
    v8f S0 = {}, S1 = {};
    {
      v16bf B0 = load_frag(Kp + (long)(kt + l16) * DK, half * 8);
      v16bf B1 = load_frag(Kp + (long)(kt + 16 + l16) * DK, half * 8);
      S0 = WMMA_BF16(QA0, B0, S0);
      S1 = WMMA_BF16(QA0, B1, S1);
      B0 = load_frag(Kp + (long)(kt + l16) * DK, 32 + half * 8);
      B1 = load_frag(Kp + (long)(kt + 16 + l16) * DK, 32 + half * 8);
      S0 = WMMA_BF16(QA1, B0, S0);
      S1 = WMMA_BF16(QA1, B1, S1);
    }

#pragma unroll
    for (int v = 0; v < 8; ++v) {
      int q = qbase + half * 8 + v;
      int k0c = kt + l16, k1c = kt + 16 + l16;
      float s0 = (k0c <= q && (q - k0c) < WIN) ? S0[v] : -1e30f;
      float s1 = (k1c <= q && (q - k1c) < WIN) ? S1[v] : -1e30f;
      float mx = fmaxf(s0, s1);
      mx = fmaxf(mx, __shfl_xor(mx, 1));
      mx = fmaxf(mx, __shfl_xor(mx, 2));
      mx = fmaxf(mx, __shfl_xor(mx, 4));
      mx = fmaxf(mx, __shfl_xor(mx, 8));
      float mnew = fmaxf(rm[v], mx);
      float corr = __expf(rm[v] - mnew);
      rm[v] = mnew;
      float p0 = (s0 > -1e29f) ? __expf(s0 - mnew) : 0.f;
      float p1 = (s1 > -1e29f) ? __expf(s1 - mnew) : 0.f;
      float ps = p0 + p1;
      ps += __shfl_xor(ps, 1);
      ps += __shfl_xor(ps, 2);
      ps += __shfl_xor(ps, 4);
      ps += __shfl_xor(ps, 8);
      rl[v] = rl[v] * corr + ps;
      O0[v] *= corr; O1[v] *= corr; O2[v] *= corr; O3[v] *= corr;
      int m = half * 8 + v;
      P[m * BPAD + l16] = (__bf16)p0;
      P[m * BPAD + 16 + l16] = (__bf16)p1;
    }

    // P (C layout in LDS) re-read as A fragment; same-wave DS ops are in order.
    v16bf PA = frag_cat(*(const v8bf*)&P[l16 * BPAD + half * 8],
                        *(const v8bf*)&P[l16 * BPAD + 16 + half * 8]);
    v16bf VB0 = load_frag(Vp + (long)(0 * 16 + l16) * TSEQ + kt, half * 8);
    v16bf VB1 = load_frag(Vp + (long)(1 * 16 + l16) * TSEQ + kt, half * 8);
    v16bf VB2 = load_frag(Vp + (long)(2 * 16 + l16) * TSEQ + kt, half * 8);
    v16bf VB3 = load_frag(Vp + (long)(3 * 16 + l16) * TSEQ + kt, half * 8);
    O0 = WMMA_BF16(PA, VB0, O0);
    O1 = WMMA_BF16(PA, VB1, O1);
    O2 = WMMA_BF16(PA, VB2, O2);
    O3 = WMMA_BF16(PA, VB3, O3);
  }

#pragma unroll
  for (int v = 0; v < 8; ++v) {
    float inv = 1.f / rl[v];
    O0[v] *= inv; O1[v] *= inv; O2[v] *= inv; O3[v] *= inv;
  }

  int bb = bh / NHEAD, h = bh % NHEAD;
  long rowbase = (long)(bb * TSEQ + qbase + half * 8);
#pragma unroll
  for (int v = 0; v < 8; ++v) {
    long rr = (rowbase + v) * DMODEL + h * DK + l16;
    AO[rr + 0]  = (__bf16)O0[v];
    AO[rr + 16] = (__bf16)O1[v];
    AO[rr + 32] = (__bf16)O2[v];
    AO[rr + 48] = (__bf16)O3[v];
  }
}

// ---------------- stage 3: output projection ----------------
__global__ void __launch_bounds__(256) k_out_gemm(
    const __bf16* __restrict__ A, const __bf16* __restrict__ wT,
    float* __restrict__ out) {
  __shared__ __align__(16) __bf16 bstage[2][64 * BPAD];

  const int NB = DMODEL / 64;  // 16 column-blocks
  int mb = blockIdx.x / NB, nb = blockIdx.x % NB;
  int m0 = mb * 128, n0 = nb * 64;
  int wave = threadIdx.x >> 5, lane = threadIdx.x & 31;
  int half = lane >> 4, l16 = lane & 15;

  int r = threadIdx.x >> 2, c = threadIdx.x & 3;
  const __bf16* gsrc = wT + (long)(n0 + r) * DMODEL + c * 8;
  unsigned soff0 = lds_off(&bstage[0][r * BPAD + c * 8]);
  unsigned soff1 = lds_off(&bstage[1][r * BPAD + c * 8]);

  const __bf16* arow = A + (long)(m0 + wave * 16 + l16) * DMODEL;

  v8f acc[4] = {};
  gemm_mainloop(arow, gsrc, &bstage[0][0], &bstage[1][0], soff0, soff1, half, l16, acc);

  long row0 = m0 + wave * 16 + half * 8;
#pragma unroll
  for (int j = 0; j < 4; ++j)
#pragma unroll
    for (int v = 0; v < 8; ++v)
      out[(row0 + v) * DMODEL + n0 + j * 16 + l16] = acc[j][v];
}

// ---------------- host ----------------
extern "C" void kernel_launch(void* const* d_in, const int* in_sizes, int n_in,
                              void* d_out, int out_size, void* d_ws, size_t ws_size,
                              hipStream_t stream) {
  const float* x    = (const float*)d_in[0];   // [2,2048,1024]
  const float* Wqkv = (const float*)d_in[1];   // [1024,3072]
  const float* Wout = (const float*)d_in[2];   // [1024,1024]
  float* out = (float*)d_out;                  // [2,2048,1024]

  char* ws = (char*)d_ws;
  size_t off = 0;
  __bf16* xb    = (__bf16*)(ws + off); off += (size_t)NTOK * DMODEL * 2;     // 8 MB
  __bf16* wqkvT = (__bf16*)(ws + off); off += (size_t)D3 * DMODEL * 2;       // 6 MB
  __bf16* woutT = (__bf16*)(ws + off); off += (size_t)DMODEL * DMODEL * 2;   // 2 MB
  __bf16* Qs    = (__bf16*)(ws + off); off += (size_t)NTOK * DK * NHEAD * 2; // 8 MB
  __bf16* Ks    = (__bf16*)(ws + off); off += (size_t)NTOK * DK * NHEAD * 2; // 8 MB
  __bf16* Vt    = (__bf16*)(ws + off); off += (size_t)NTOK * DK * NHEAD * 2; // 8 MB
  __bf16* AO    = (__bf16*)(ws + off); off += (size_t)NTOK * DMODEL * 2;     // 8 MB

  (void)in_sizes; (void)n_in; (void)out_size; (void)ws_size;

  k_cvt_x<<<(NTOK * DMODEL) / 256, 256, 0, stream>>>(x, xb, NTOK * DMODEL);
  k_transpose<<<(DMODEL * D3) / 256, 256, 0, stream>>>(Wqkv, wqkvT, DMODEL, D3);
  k_transpose<<<(DMODEL * DMODEL) / 256, 256, 0, stream>>>(Wout, woutT, DMODEL, DMODEL);

  // (NTOK/128) x (D3/64) = 32 x 48 = 1536 blocks
  k_qkv_gemm<<<1536, 256, 0, stream>>>(xb, wqkvT, Qs, Ks, Vt);

  // B*H*(T/16) = 4096 q-tiles / 8 waves per block
  k_attn<<<4096 / 8, 256, 0, stream>>>(Qs, Ks, Vt, AO);

  // (NTOK/128) x (DMODEL/64) = 32 x 16 = 512 blocks
  k_out_gemm<<<512, 256, 0, stream>>>(AO, woutT, out);
}
